// CompoundGCN_75067438399773
// MI455X (gfx1250) — compile-verified, matrix operand order checked by reading, not run on hardware
//
#include <hip/hip_runtime.h>
#include <cstdint>
#include <cstddef>

// ---------------------------------------------------------------------------
// CompoundGCN forward for MI455X (gfx1250, wave32, WMMA).
//   N=50000 nodes, E=600000 edges, D=64, HID=256, OUT=1, B=64, 2 msg rounds.
// Dense transforms on v_wmma_f32_16x16x32_bf16 (f32 accumulate).
// Weights are pre-packed once per launch into fragment-ready bf16 layout so
// every B-fragment is one 32B vector load; A-fragments are b128 loads + cvt.
// Scatter/gather via L2 atomics; pooling in LDS.
// ---------------------------------------------------------------------------

#define N_NODES 50000
#define N_EDGES 600000
#define DIM     64
#define HIDDEN  256
#define NGRAPH  64
#define MAXDEG  10
#define TPB     256
#define RPW     8        // row tiles per wave in k_gemm

typedef __attribute__((ext_vector_type(16))) __bf16 v16bf;
typedef __attribute__((ext_vector_type(8)))  float  v8f;

__device__ __forceinline__ v8f wmma_bf16(v16bf a, v16bf b, v8f c) {
  // 8 args: (neg_a, A, neg_b, B, c_mod, C, reuse_a, reuse_b)
  return __builtin_amdgcn_wmma_f32_16x16x32_bf16(false, a, false, b,
                                                 (short)0, c, false, false);
}

__device__ __forceinline__ void cvt8(v16bf& a, int base, float4 x0, float4 x1) {
  a[base + 0] = (__bf16)x0.x; a[base + 1] = (__bf16)x0.y;
  a[base + 2] = (__bf16)x0.z; a[base + 3] = (__bf16)x0.w;
  a[base + 4] = (__bf16)x1.x; a[base + 5] = (__bf16)x1.y;
  a[base + 6] = (__bf16)x1.z; a[base + 7] = (__bf16)x1.w;
}

// A fragment, 16x32 bf16, contiguous f32 rows; 32B-aligned pointers.
// lane&15 = row(M), kb = (lane&16)?8:0; K elems {kb..kb+7, 16+kb..23+kb}.
__device__ __forceinline__ v16bf load_frag_rm(const float* __restrict__ p, int kb) {
  const float4* q0 = (const float4*)(p + kb);
  const float4* q1 = (const float4*)(p + 16 + kb);
  v16bf a;
  cvt8(a, 0, q0[0], q0[1]);
  cvt8(a, 8, q1[0], q1[1]);
  return a;
}

// Pre-packed B fragment: one 32B vector load per lane.
__device__ __forceinline__ v16bf load_frag_packed(const __bf16* __restrict__ P,
                                                  int frag, int lane) {
  return *(const v16bf*)(P + (((size_t)frag * 32 + lane) << 4));
}

// order-preserving float<->uint encoding for atomic max on floats
__device__ __forceinline__ unsigned f2ord(float f) {
  unsigned u = __float_as_uint(f);
  return (u & 0x80000000u) ? ~u : (u | 0x80000000u);
}
__device__ __forceinline__ float ord2f(unsigned u) {
  return __uint_as_float((u & 0x80000000u) ? (u & 0x7FFFFFFFu) : ~u);
}
#define ORD_NEG_INF 0x007FFFFFu   // f2ord(-inf)

// ---------------------------------------------------------------------------
// weight pre-pack: P[(((ks*nt + ct)*32 + lane)*16 + elem)] = bf16(B(k,c))
//   B(k,c) = trans ? W[c*K + k] : W[k*Dout + c];  nmat matrices back-to-back.
// ---------------------------------------------------------------------------
__global__ void k_pack_w(const float* __restrict__ W, __bf16* __restrict__ P,
                         int K, int Dout, int trans, int nmat) {
  long total = (long)nmat * K * Dout;
  long idx = (long)blockIdx.x * blockDim.x + threadIdx.x;
  if (idx >= total) return;
  int per = K * Dout;
  int mtx = (int)(idx / per);
  int r   = (int)(idx % per);
  int elem = r & 15;
  int lane = (r >> 4) & 31;
  int f    = r >> 9;
  int nt = Dout >> 4;
  int ct = f % nt;
  int ks = f / nt;
  int ln15 = lane & 15;
  int kb = (lane & 16) ? 8 : 0;
  int kl = (elem < 8) ? (kb + elem) : (16 + kb + elem - 8);
  int k = ks * 32 + kl;
  int c = ct * 16 + ln15;
  const float* Wm = W + (size_t)mtx * per;
  float v = trans ? Wm[(size_t)c * K + k] : Wm[(size_t)k * Dout + c];
  P[idx] = (__bf16)v;
}

// ---------------------------------------------------------------------------
// utility fill
// ---------------------------------------------------------------------------
__global__ void k_fill_u32(unsigned* __restrict__ p, size_t n, unsigned v) {
  size_t i = (size_t)blockIdx.x * blockDim.x + threadIdx.x;
  size_t s = (size_t)gridDim.x * blockDim.x;
  for (; i < n; i += s) p[i] = v;
}

// ---------------------------------------------------------------------------
// degree / binning
// ---------------------------------------------------------------------------
__global__ void k_count_deg(const int* __restrict__ colI, int* __restrict__ cnt) {
  int e = blockIdx.x * blockDim.x + threadIdx.x;
  if (e < N_EDGES) atomicAdd(&cnt[colI[e]], 1);
}

__global__ void k_make_deg(const int* __restrict__ cnt, int* __restrict__ deg,
                           int* __restrict__ hasin, int* __restrict__ bin_cnt) {
  int n = blockIdx.x * blockDim.x + threadIdx.x;
  if (n >= N_NODES) return;
  int c = cnt[n];
  int d = c > MAXDEG ? MAXDEG : c;
  deg[n] = d;
  hasin[n] = (c > 0) ? 1 : 0;
  atomicAdd(&bin_cnt[d], 1);
}

__global__ void k_scan_bins(const int* __restrict__ bin_cnt,
                            int* __restrict__ bin_off, int* __restrict__ poff) {
  if (blockIdx.x == 0 && threadIdx.x == 0) {
    int o = 0, p = 0;
    for (int b = 0; b <= MAXDEG; ++b) {
      bin_off[b] = o; poff[b] = p;
      o += bin_cnt[b];
      p += (bin_cnt[b] + 15) & ~15;
    }
    bin_off[MAXDEG + 1] = o;
    poff[MAXDEG + 1] = p;
  }
}

__global__ void k_fill_perm(const int* __restrict__ deg, const int* __restrict__ bin_off,
                            int* __restrict__ bin_fill, int* __restrict__ perm) {
  int n = blockIdx.x * blockDim.x + threadIdx.x;
  if (n >= N_NODES) return;
  int d = deg[n];
  int p = bin_off[d] + atomicAdd(&bin_fill[d], 1);
  perm[p] = n;
}

// ---------------------------------------------------------------------------
// WMMA GEMM: C[M,Dout] = act(A[M,K] @ B + bias), K = KSTEPS*32, B pre-packed.
// Each wave owns one 16-wide column tile; B-fragments (all K steps) live in
// registers and are reused across RPW row tiles. All exits are wave-uniform.
// ---------------------------------------------------------------------------
template <int RELU, int KSTEPS>
__global__ void k_gemm(const float* __restrict__ A, const __bf16* __restrict__ Wpk,
                       const float* __restrict__ bias, float* __restrict__ C,
                       int M, int Dout) {
  const int K = KSTEPS * 32;
  int w    = (blockIdx.x * blockDim.x + threadIdx.x) >> 5;
  int lane = threadIdx.x & 31;
  int nt = Dout >> 4;
  int mt = (M + 15) >> 4;
  int rowWaves = (mt + RPW - 1) / RPW;
  if (w >= nt * rowWaves) return;                 // wave-uniform
  int ct  = w % nt;
  int n0  = ct << 4;
  int rt0 = (w / nt) * RPW;
  int ln15 = lane & 15;
  int kb   = (lane & 16) ? 8 : 0;

  v16bf bf[KSTEPS];
#pragma unroll
  for (int ks = 0; ks < KSTEPS; ++ks)
    bf[ks] = load_frag_packed(Wpk, ks * nt + ct, lane);

  float bv = bias ? bias[n0 + ln15] : 0.f;
  int rb = (lane & 16) ? 8 : 0;

  for (int r = 0; r < RPW; ++r) {
    int m0 = (rt0 + r) << 4;
    if (m0 >= M) return;                          // wave-uniform
    v8f acc;
#pragma unroll
    for (int j = 0; j < 8; ++j) acc[j] = bv;
#pragma unroll
    for (int ks = 0; ks < KSTEPS; ++ks) {
      v16bf a = load_frag_rm(A + (size_t)(m0 + ln15) * K + ks * 32, kb);
      acc = wmma_bf16(a, bf[ks], acc);
    }
#pragma unroll
    for (int j = 0; j < 8; ++j) {
      float v = acc[j];
      if (RELU) v = fmaxf(v, 0.f);
      C[(size_t)(m0 + rb + j) * Dout + n0 + ln15] = v;
    }
  }
}

// ---------------------------------------------------------------------------
// scatter-add: agg[col[e]] += nodeF[row[e]]   (thread = edge x 4-dim chunk)
// ---------------------------------------------------------------------------
__global__ void k_scatter_add(const int* __restrict__ rowI, const int* __restrict__ colI,
                              const float* __restrict__ nodeF, float* __restrict__ agg) {
  int idx = blockIdx.x * blockDim.x + threadIdx.x;
  if (idx >= N_EDGES * 16) return;
  int e = idx >> 4;
  int q = (idx & 15) << 2;
  const float* s = nodeF + (size_t)rowI[e] * DIM + q;
  float* d       = agg   + (size_t)colI[e] * DIM + q;
  atomicAdd(&d[0], s[0]);
  atomicAdd(&d[1], s[1]);
  atomicAdd(&d[2], s[2]);
  atomicAdd(&d[3], s[3]);
}

// ---------------------------------------------------------------------------
// MFConv, degree-binned: m[n] = relu(agg[n] @ Wl[deg] + bl[deg] + nodeF[n] @ Wr[deg])
// Nodes permuted into contiguous degree bins; bins padded to 16 rows.
// Padding rows read node 0 (finite) unmasked — WMMA D-row m depends only on
// A-row m, so garbage stays in padding rows, which are never stored.
// ---------------------------------------------------------------------------
__global__ void k_mfconv(const float* __restrict__ agg, const float* __restrict__ nodeF,
                         const __bf16* __restrict__ WlP, const float* __restrict__ bl,
                         const __bf16* __restrict__ WrP,
                         const int* __restrict__ perm, const int* __restrict__ bin_off,
                         const int* __restrict__ poff, float* __restrict__ m) {
  int w    = (blockIdx.x * blockDim.x + threadIdx.x) >> 5;
  int lane = threadIdx.x & 31;
  int rt = w >> 2;        // padded row tile
  int ct = w & 3;         // column tile (DIM/16 = 4)
  int r0 = rt << 4;
  if (r0 >= poff[MAXDEG + 1]) return;     // wave-uniform
  int b = 0;
  while (r0 >= poff[b + 1]) ++b;
  int base = r0 - poff[b];
  int p0   = bin_off[b] + base;
  int cntb = bin_off[b + 1] - bin_off[b];

  int ln15 = lane & 15;
  int kb   = (lane & 16) ? 8 : 0;
  int n0   = ct << 4;

  int node = ((base + ln15) < cntb) ? perm[p0 + ln15] : 0;

  const __bf16* WlB = WlP + (size_t)b * DIM * DIM;   // packed, per-matrix stride K*Dout
  const __bf16* WrB = WrP + (size_t)b * DIM * DIM;

  float bv = bl[b * DIM + n0 + ln15];
  v8f acc;
#pragma unroll
  for (int j = 0; j < 8; ++j) acc[j] = bv;

#pragma unroll
  for (int ks = 0; ks < 2; ++ks) {
    int k0 = ks * 32;
    v16bf a  = load_frag_rm(agg + (size_t)node * DIM + k0, kb);
    v16bf bm = load_frag_packed(WlB, ks * 4 + ct, lane);
    acc = wmma_bf16(a, bm, acc);
    a  = load_frag_rm(nodeF + (size_t)node * DIM + k0, kb);
    bm = load_frag_packed(WrB, ks * 4 + ct, lane);
    acc = wmma_bf16(a, bm, acc);
  }

  int rb = (lane & 16) ? 8 : 0;
#pragma unroll
  for (int j = 0; j < 8; ++j) {
    int rr = base + rb + j;
    if (rr < cntb) {
      int nd = perm[p0 + rb + j];
      m[(size_t)nd * DIM + n0 + ln15] = fmaxf(acc[j], 0.f);
    }
  }
}

// ---------------------------------------------------------------------------
// GRU elementwise: h = (1-z)*n + z*h   (gates from gi, gh [N, 3D])
// ---------------------------------------------------------------------------
__global__ void k_gru(const float* __restrict__ gi, const float* __restrict__ gh,
                      float* __restrict__ h) {
  int idx = blockIdx.x * blockDim.x + threadIdx.x;
  if (idx >= N_NODES * DIM) return;
  int n = idx >> 6, d = idx & 63;
  const float* gin = gi + (size_t)n * (3 * DIM);
  const float* ghn = gh + (size_t)n * (3 * DIM);
  float r  = 1.f / (1.f + expf(-(gin[d] + ghn[d])));
  float z  = 1.f / (1.f + expf(-(gin[DIM + d] + ghn[DIM + d])));
  float nn = tanhf(gin[2 * DIM + d] + r * ghn[2 * DIM + d]);
  size_t o = (size_t)n * DIM + d;
  float hv = h[o];
  h[o] = (1.f - z) * nn + z * hv;
}

// ---------------------------------------------------------------------------
// EdgeConv fused: msg = [xi, xj-xi] @ ec_w + ec_b ; atomic segment-max on col.
// EA is the N x 64 edge-feature state (only rows < N are ever referenced).
// Each 8-wide K chunk lies entirely on one side of the 64 boundary -> two
// b128 loads per chunk (plus a subtract on the high half).
// ---------------------------------------------------------------------------
__device__ __forceinline__ void ec_chunk(v16bf& a, int base, int kk,
                                         const float* __restrict__ xi,
                                         const float* __restrict__ xj) {
  if (kk < DIM) {
    const float4* q = (const float4*)(xi + kk);
    cvt8(a, base, q[0], q[1]);
  } else {
    int k2 = kk - DIM;
    const float4* qi = (const float4*)(xi + k2);
    const float4* qj = (const float4*)(xj + k2);
    float4 i0 = qi[0], i1 = qi[1], j0 = qj[0], j1 = qj[1];
    float4 d0, d1;
    d0.x = j0.x - i0.x; d0.y = j0.y - i0.y; d0.z = j0.z - i0.z; d0.w = j0.w - i0.w;
    d1.x = j1.x - i1.x; d1.y = j1.y - i1.y; d1.z = j1.z - i1.z; d1.w = j1.w - i1.w;
    cvt8(a, base, d0, d1);
  }
}

__global__ void k_edgeconv(const float* __restrict__ EA, const int* __restrict__ rowI,
                           const int* __restrict__ colI, const __bf16* __restrict__ ecP,
                           const float* __restrict__ ecb, unsigned* __restrict__ aggEnc) {
  int w    = (blockIdx.x * blockDim.x + threadIdx.x) >> 5;
  int lane = threadIdx.x & 31;
  int et = w >> 2;
  int ct = w & 3;
  if (et >= N_EDGES / 16) return;          // wave-uniform
  int ln15 = lane & 15;
  int kb   = (lane & 16) ? 8 : 0;
  int n0   = ct << 4;

  int e  = et * 16 + ln15;
  int ci = colI[e], ri = rowI[e];
  const float* xi = EA + (size_t)ci * DIM;
  const float* xj = EA + (size_t)ri * DIM;

  float bv = ecb[n0 + ln15];
  v8f acc;
#pragma unroll
  for (int j = 0; j < 8; ++j) acc[j] = bv;

#pragma unroll
  for (int ks = 0; ks < 4; ++ks) {
    int k0 = ks * 32;
    v16bf a;
    ec_chunk(a, 0, k0 + kb, xi, xj);
    ec_chunk(a, 8, k0 + 16 + kb, xi, xj);
    v16bf b = load_frag_packed(ecP, ks * 4 + ct, lane);
    acc = wmma_bf16(a, b, acc);
  }

  int rb = (lane & 16) ? 8 : 0;
#pragma unroll
  for (int j = 0; j < 8; ++j) {
    int ej = et * 16 + rb + j;
    int cj = colI[ej];
    atomicMax(&aggEnc[(size_t)cj * DIM + n0 + ln15], f2ord(acc[j]));
  }
}

__global__ void k_edge_finish(const unsigned* __restrict__ aggEnc,
                              const int* __restrict__ hasin,
                              float* __restrict__ EAnext,
                              float* __restrict__ embEdge /* nullable */) {
  int idx = blockIdx.x * blockDim.x + threadIdx.x;
  if (idx >= N_NODES * DIM) return;
  int n = idx >> 6;
  float v = hasin[n] ? ord2f(aggEnc[idx]) : 0.f;
  if (embEdge) embEdge[idx] = v;
  EAnext[idx] = fmaxf(v, 0.f);
}

// ---------------------------------------------------------------------------
// readout pooling: g[batch[col[e]]] += [nodeF[row[e]], EA[col[e]]]
// Full 64x128 accumulator lives in 32KB LDS per workgroup.
// ---------------------------------------------------------------------------
__global__ void k_pool(const float* __restrict__ nodeF, const float* __restrict__ EA,
                       const int* __restrict__ rowI, const int* __restrict__ colI,
                       const int* __restrict__ batch, float* __restrict__ g) {
  __shared__ float sg[NGRAPH * 2 * DIM];   // 8192 floats = 32 KB
  for (int i = threadIdx.x; i < NGRAPH * 2 * DIM; i += blockDim.x) sg[i] = 0.f;
  __syncthreads();

  long total = (long)N_EDGES * 32;
  for (long idx = (long)blockIdx.x * blockDim.x + threadIdx.x; idx < total;
       idx += (long)gridDim.x * blockDim.x) {
    int e = (int)(idx >> 5);
    int q = ((int)idx & 31) << 2;          // dim chunk of 4, 0..124
    int ci = colI[e];
    int b  = batch[ci];
    float* dst = sg + b * (2 * DIM) + q;
    const float* s = (q < DIM) ? (nodeF + (size_t)rowI[e] * DIM + q)
                               : (EA    + (size_t)ci      * DIM + (q - DIM));
    atomicAdd(&dst[0], s[0]);
    atomicAdd(&dst[1], s[1]);
    atomicAdd(&dst[2], s[2]);
    atomicAdd(&dst[3], s[3]);
  }
  __syncthreads();
  for (int i = threadIdx.x; i < NGRAPH * 2 * DIM; i += blockDim.x)
    atomicAdd(&g[i], sg[i]);
}

__global__ void k_p2(const float* __restrict__ g2, const float* __restrict__ w,
                     const float* __restrict__ b, float* __restrict__ out) {
  int i = blockIdx.x * blockDim.x + threadIdx.x;
  if (i >= NGRAPH) return;
  float s = b[0];
  for (int k = 0; k < HIDDEN; ++k) s += g2[(size_t)i * HIDDEN + k] * w[k];
  out[i] = s;
}

// ---------------------------------------------------------------------------
// host launcher
// ---------------------------------------------------------------------------
static inline int blocksFor(long long work, int tpb) {
  return (int)((work + tpb - 1) / tpb);
}
static inline int gemmBlocks(int M, int Dout) {
  int mt = (M + 15) / 16;
  int waves = (Dout / 16) * ((mt + RPW - 1) / RPW);
  return (waves * 32 + TPB - 1) / TPB;
}

extern "C" void kernel_launch(void* const* d_in, const int* in_sizes, int n_in,
                              void* d_out, int out_size, void* d_ws, size_t ws_size,
                              hipStream_t stream) {
  (void)in_sizes; (void)n_in; (void)out_size; (void)ws_size;

  const float* x        = (const float*)d_in[0];
  const float* edgeAttr = (const float*)d_in[1];
  const int*   eIdx     = (const int*)  d_in[2];
  const int*   batch    = (const int*)  d_in[3];
  const float* lin0_w   = (const float*)d_in[4];
  const float* lin0_b   = (const float*)d_in[5];
  const float* mf_Wl    = (const float*)d_in[6];
  const float* mf_bl    = (const float*)d_in[7];
  const float* mf_Wr    = (const float*)d_in[8];
  const float* ec_w     = (const float*)d_in[9];
  const float* ec_b     = (const float*)d_in[10];
  const float* gru_wih  = (const float*)d_in[11];
  const float* gru_whh  = (const float*)d_in[12];
  const float* gru_bih  = (const float*)d_in[13];
  const float* gru_bhh  = (const float*)d_in[14];
  const float* p0_w     = (const float*)d_in[15];
  const float* p0_b     = (const float*)d_in[16];
  const float* p1_w     = (const float*)d_in[17];
  const float* p1_b     = (const float*)d_in[18];
  const float* p2_w     = (const float*)d_in[19];
  const float* p2_b     = (const float*)d_in[20];

  const int* rowI = eIdx;              // edge_index[0]
  const int* colI = eIdx + N_EDGES;    // edge_index[1]

  // output layout: g [64] | emb_node [N*64] | emb_edge [E*64]
  float* out_g    = (float*)d_out;
  float* out_node = out_g + NGRAPH;
  float* out_edge = out_node + (size_t)N_NODES * DIM;

  // workspace carve-up (256B aligned)
  char* ws = (char*)d_ws;
  size_t off = 0;
  auto carve = [&](size_t bytes) -> char* {
    char* p = ws + off;
    off = (off + bytes + 255) & ~(size_t)255;
    return p;
  };
  float*    nodeF   = (float*)   carve((size_t)N_NODES * DIM * 4);
  float*    agg     = (float*)   carve((size_t)N_NODES * DIM * 4);
  float*    m0      = (float*)   carve((size_t)N_NODES * DIM * 4);
  float*    gi      = (float*)   carve((size_t)N_NODES * 3 * DIM * 4);
  float*    gh      = (float*)   carve((size_t)N_NODES * 3 * DIM * 4);
  float*    EAbuf   = (float*)   carve((size_t)N_NODES * DIM * 4);
  unsigned* aggEnc  = (unsigned*)carve((size_t)N_NODES * DIM * 4);
  int*      degCnt  = (int*)     carve((size_t)N_NODES * 4);
  int*      deg     = (int*)     carve((size_t)N_NODES * 4);
  int*      hasin   = (int*)     carve((size_t)N_NODES * 4);
  int*      perm    = (int*)     carve((size_t)N_NODES * 4);
  int*      binCnt  = (int*)     carve(16 * 4);
  int*      binOff  = (int*)     carve(16 * 4);
  int*      binFill = (int*)     carve(16 * 4);
  int*      poff    = (int*)     carve(16 * 4);
  float*    gpool   = (float*)   carve((size_t)NGRAPH * 2 * DIM * 4);
  float*    g1      = (float*)   carve((size_t)NGRAPH * HIDDEN * 4);
  float*    g2      = (float*)   carve((size_t)NGRAPH * HIDDEN * 4);
  // packed bf16 weights (fragment-ready)
  __bf16*   lin0P   = (__bf16*)  carve((size_t)DIM * DIM * 2);
  __bf16*   wihP    = (__bf16*)  carve((size_t)DIM * 3 * DIM * 2);
  __bf16*   whhP    = (__bf16*)  carve((size_t)DIM * 3 * DIM * 2);
  __bf16*   WlPk    = (__bf16*)  carve((size_t)(MAXDEG + 1) * DIM * DIM * 2);
  __bf16*   WrPk    = (__bf16*)  carve((size_t)(MAXDEG + 1) * DIM * DIM * 2);
  __bf16*   ecPk    = (__bf16*)  carve((size_t)2 * DIM * DIM * 2);
  __bf16*   p0Pk    = (__bf16*)  carve((size_t)2 * DIM * HIDDEN * 2);
  __bf16*   p1Pk    = (__bf16*)  carve((size_t)HIDDEN * HIDDEN * 2);

  // ---- pre-pack weights into fragment-ready bf16 (once per launch, ~440KB)
  k_pack_w<<<blocksFor(DIM * DIM, TPB), TPB, 0, stream>>>(lin0_w, lin0P, DIM, DIM, 0, 1);
  k_pack_w<<<blocksFor(DIM * 3 * DIM, TPB), TPB, 0, stream>>>(gru_wih, wihP, DIM, 3 * DIM, 1, 1);
  k_pack_w<<<blocksFor(DIM * 3 * DIM, TPB), TPB, 0, stream>>>(gru_whh, whhP, DIM, 3 * DIM, 1, 1);
  k_pack_w<<<blocksFor((MAXDEG + 1) * DIM * DIM, TPB), TPB, 0, stream>>>(
      mf_Wl, WlPk, DIM, DIM, 0, MAXDEG + 1);
  k_pack_w<<<blocksFor((MAXDEG + 1) * DIM * DIM, TPB), TPB, 0, stream>>>(
      mf_Wr, WrPk, DIM, DIM, 0, MAXDEG + 1);
  k_pack_w<<<blocksFor(2 * DIM * DIM, TPB), TPB, 0, stream>>>(ec_w, ecPk, 2 * DIM, DIM, 0, 1);
  k_pack_w<<<blocksFor(2 * DIM * HIDDEN, TPB), TPB, 0, stream>>>(p0_w, p0Pk, 2 * DIM, HIDDEN, 0, 1);
  k_pack_w<<<blocksFor(HIDDEN * HIDDEN, TPB), TPB, 0, stream>>>(p1_w, p1Pk, HIDDEN, HIDDEN, 0, 1);

  // ---- degree counts + degree bins
  k_fill_u32<<<256, TPB, 0, stream>>>((unsigned*)degCnt, N_NODES, 0u);
  k_fill_u32<<<1, 64, 0, stream>>>((unsigned*)binCnt, 16, 0u);
  k_fill_u32<<<1, 64, 0, stream>>>((unsigned*)binFill, 16, 0u);
  k_count_deg<<<blocksFor(N_EDGES, TPB), TPB, 0, stream>>>(colI, degCnt);
  k_make_deg<<<blocksFor(N_NODES, TPB), TPB, 0, stream>>>(degCnt, deg, hasin, binCnt);
  k_scan_bins<<<1, 1, 0, stream>>>(binCnt, binOff, poff);
  k_fill_perm<<<blocksFor(N_NODES, TPB), TPB, 0, stream>>>(deg, binOff, binFill, perm);

  // ---- zero tail of emb_edge (rows N..E-1 are always zero)
  k_fill_u32<<<2048, TPB, 0, stream>>>((unsigned*)(out_edge + (size_t)N_NODES * DIM),
                                       (size_t)(N_EDGES - N_NODES) * DIM, 0u);

  // ---- lin0: nodeF = relu(x @ lin0_w + b)   (K=64 -> KSTEPS=2)
  k_gemm<1, 2><<<gemmBlocks(N_NODES, DIM), TPB, 0, stream>>>(
      x, lin0P, lin0_b, nodeF, N_NODES, DIM);

  const int mfTilesMax = (N_NODES / 16 + MAXDEG + 1);            // padded bound
  const int mfBlocks   = (mfTilesMax * 4 * 32 + TPB - 1) / TPB;
  const int ecBlocks   = ((N_EDGES / 16) * 4 * 32 + TPB - 1) / TPB;

  for (int it = 0; it < 2; ++it) {
    const float* EAcur = (it == 0) ? edgeAttr : EAbuf;   // only rows <N read
    float* mdst  = (it == 1) ? out_node : m0;            // emb_node = last m
    float* embE  = (it == 1) ? out_edge : nullptr;       // emb_edge = last edge_new

    // agg = segment_sum(nodeF[row], col)
    k_fill_u32<<<1024, TPB, 0, stream>>>((unsigned*)agg, (size_t)N_NODES * DIM, 0u);
    k_scatter_add<<<blocksFor((long long)N_EDGES * 16, TPB), TPB, 0, stream>>>(
        rowI, colI, nodeF, agg);

    // MFConv (degree-binned, one bin per node)
    k_mfconv<<<mfBlocks, TPB, 0, stream>>>(agg, nodeF, WlPk, mf_bl, WrPk,
                                           perm, binOff, poff, mdst);

    // GRU: gi = m @ Wih^T + bih ; gh = h @ Whh^T + bhh ; h update in place
    k_gemm<0, 2><<<gemmBlocks(N_NODES, 3 * DIM), TPB, 0, stream>>>(
        mdst, wihP, gru_bih, gi, N_NODES, 3 * DIM);
    k_gemm<0, 2><<<gemmBlocks(N_NODES, 3 * DIM), TPB, 0, stream>>>(
        nodeF, whhP, gru_bhh, gh, N_NODES, 3 * DIM);
    k_gru<<<blocksFor((long long)N_NODES * DIM, TPB), TPB, 0, stream>>>(gi, gh, nodeF);

    // EdgeConv + fused segment max
    k_fill_u32<<<1024, TPB, 0, stream>>>(aggEnc, (size_t)N_NODES * DIM, ORD_NEG_INF);
    k_edgeconv<<<ecBlocks, TPB, 0, stream>>>(EAcur, rowI, colI, ecPk, ec_b, aggEnc);
    k_edge_finish<<<blocksFor((long long)N_NODES * DIM, TPB), TPB, 0, stream>>>(
        aggEnc, hasin, EAbuf, embE);
  }

  // ---- readout: pool edges -> per-graph vector -> MLP
  k_fill_u32<<<32, TPB, 0, stream>>>((unsigned*)gpool, (size_t)NGRAPH * 2 * DIM, 0u);
  k_pool<<<1024, TPB, 0, stream>>>(nodeF, EAbuf, rowI, colI, batch, gpool);
  k_gemm<1, 4><<<gemmBlocks(NGRAPH, HIDDEN), TPB, 0, stream>>>(
      gpool, p0Pk, p0_b, g1, NGRAPH, HIDDEN);           // K=128
  k_gemm<1, 8><<<gemmBlocks(NGRAPH, HIDDEN), TPB, 0, stream>>>(
      g1, p1Pk, p1_b, g2, NGRAPH, HIDDEN);              // K=256
  k_p2<<<1, NGRAPH, 0, stream>>>(g2, p2_w, p2_b, out_g);
}
